// Attention_30056181138106
// MI455X (gfx1250) — matrix operand order repaired and checked by reading, not compiled, not run on hardware
//
#include <hip/hip_runtime.h>
#include <hip/hip_bf16.h>

typedef __attribute__((ext_vector_type(16))) _Float16 v16h;
typedef __attribute__((ext_vector_type(8)))  _Float16 v8h;
typedef __attribute__((ext_vector_type(8)))  float    v8f;
typedef __attribute__((ext_vector_type(4)))  float    v4f;

#define NB   8
#define NL   2048
#define DM   1024
#define DH   128
#define TAU  30.0f
#define SCALE 0.08838834764831845f  /* 1/sqrt(128) */

// ---------------------------------------------------------------------------
// Projection: Q = x @ W_Q^T, K = x @ W_K^T, V = x @ W_V^T  (f32 -> f16, WMMA)
// Grid: (B*L/16)*3 blocks, 256 threads (8 waves). Each wave: one 16x16 tile.
// Q,K stored row-major [B*L][128] f16. V stored transposed [B][128][L] f16.
// ---------------------------------------------------------------------------
__global__ __launch_bounds__(256) void proj_kernel(
    const float* __restrict__ x,
    const float* __restrict__ WQ,
    const float* __restrict__ WK,
    const float* __restrict__ WV,
    _Float16* __restrict__ Qb,
    _Float16* __restrict__ Kb,
    _Float16* __restrict__ Vt)
{
    const int lane = threadIdx.x & 31;
    const int wid  = threadIdx.x >> 5;
    const int l16  = lane & 15;
    const int half = lane >> 4;

    const int proj    = blockIdx.x % 3;
    const int mt      = blockIdx.x / 3;        // 0 .. B*L/16-1
    const int rowbase = mt * 16;
    const int colbase = wid * 16;              // 8 waves cover 128 cols

    const float* W = (proj == 0) ? WQ : (proj == 1) ? WK : WV;

    const float* xrow = x + (size_t)(rowbase + l16) * DM;   // A-matrix row
    const float* wrow = W + (size_t)(colbase + l16) * DM;   // B-matrix column

    v8f acc = {};
    for (int kb = 0; kb < DM; kb += 32) {
        // A fragment 16x32: lane<16 -> K 0..15; lane>=16 -> K 16..31
        const float* ap = xrow + kb + half * 16;
        v4f f0 = *(const v4f*)(ap);
        v4f f1 = *(const v4f*)(ap + 4);
        v4f f2 = *(const v4f*)(ap + 8);
        v4f f3 = *(const v4f*)(ap + 12);
        v16h a;
        #pragma unroll
        for (int i = 0; i < 4; ++i) {
            a[i]      = (_Float16)f0[i];
            a[4 + i]  = (_Float16)f1[i];
            a[8 + i]  = (_Float16)f2[i];
            a[12 + i] = (_Float16)f3[i];
        }
        // B fragment 32x16: lane<16 -> K 0..15 of column l16; lane>=16 -> K 16..31
        const float* bp = wrow + kb + half * 16;
        v4f g0 = *(const v4f*)(bp);
        v4f g1 = *(const v4f*)(bp + 4);
        v4f g2 = *(const v4f*)(bp + 8);
        v4f g3 = *(const v4f*)(bp + 12);
        v16h bfrag;
        #pragma unroll
        for (int i = 0; i < 4; ++i) {
            bfrag[i]      = (_Float16)g0[i];
            bfrag[4 + i]  = (_Float16)g1[i];
            bfrag[8 + i]  = (_Float16)g2[i];
            bfrag[12 + i] = (_Float16)g3[i];
        }
        acc = __builtin_amdgcn_wmma_f32_16x16x32_f16(
            false, a, false, bfrag, (short)0, acc, false, false);
    }

    // C layout: lanes 0-15 -> M=r, lanes 16-31 -> M=r+8; N = l16
    if (proj < 2) {
        _Float16* dst = (proj == 0 ? Qb : Kb);
        #pragma unroll
        for (int r = 0; r < 8; ++r) {
            int grow = rowbase + r + half * 8;
            dst[(size_t)grow * DH + colbase + l16] = (_Float16)acc[r];
        }
    } else {
        #pragma unroll
        for (int r = 0; r < 8; ++r) {
            int grow = rowbase + r + half * 8;
            int b = grow / NL;
            int l = grow % NL;
            Vt[((size_t)b * DH + colbase + l16) * NL + l] = (_Float16)acc[r];
        }
    }
}

// ---------------------------------------------------------------------------
// Flash attention with tanh soft-capping, causal + padding mask.
// One wave = one 16-query tile. 4 waves per block (256 blocks). Keys in
// chunks of 32. K/V B-fragments are contiguous 32B per lane -> direct v16h.
// ---------------------------------------------------------------------------
__global__ __launch_bounds__(128) void attn_kernel(
    const _Float16* __restrict__ Qb,
    const _Float16* __restrict__ Kb,
    const _Float16* __restrict__ Vt,
    const int*      __restrict__ amask,
    float*          __restrict__ out)
{
    __shared__ _Float16 lds_p[4][16 * 32];   // per-wave P tile (16 q x 32 k)

    const int lane = threadIdx.x & 31;
    const int wid  = threadIdx.x >> 5;
    const int l16  = lane & 15;
    const int half = lane >> 4;

    const int gw    = blockIdx.x * 4 + wid;  // 0 .. B*L/16-1
    const int qt    = gw % (NL / 16);
    const int b     = gw / (NL / 16);
    const int qbase = qt * 16;

    // Preload Q A-fragments for the 4 d-chunks of 32
    v16h aq[4];
    {
        const _Float16* qp = Qb + ((size_t)b * NL + qbase + l16) * DH;
        #pragma unroll
        for (int c = 0; c < 4; ++c) {
            const _Float16* p = qp + c * 32 + half * 16;
            v8h lo = *(const v8h*)(p);
            v8h hi = *(const v8h*)(p + 8);
            #pragma unroll
            for (int i = 0; i < 8; ++i) { aq[c][i] = lo[i]; aq[c][8 + i] = hi[i]; }
        }
    }

    float m_run[8], l_run[8];
    v8f   o[8];
    #pragma unroll
    for (int r = 0; r < 8; ++r) { m_run[r] = -1e30f; l_run[r] = 0.0f; }
    #pragma unroll
    for (int c = 0; c < 8; ++c) o[c] = (v8f){};

    const int nk32 = (qt + 2) >> 1;          // ceil((qt+1)/2) key chunks of 32
    for (int kc = 0; kc < nk32; ++kc) {
        const int kbase = kc * 32;

        // --- S = Q @ K^T for two 16-key tiles ---
        v8f s0 = {}, s1 = {};
        const _Float16* krow = Kb + ((size_t)b * NL + kbase + l16) * DH + half * 16;
        #pragma unroll
        for (int c = 0; c < 4; ++c) {
            // B fragment: 16 contiguous halves per lane (32B aligned)
            v16h bk0 = *(const v16h*)(krow + c * 32);
            s0 = __builtin_amdgcn_wmma_f32_16x16x32_f16(
                false, aq[c], false, bk0, (short)0, s0, false, false);
            v16h bk1 = *(const v16h*)(krow + (size_t)16 * DH + c * 32);
            s1 = __builtin_amdgcn_wmma_f32_16x16x32_f16(
                false, aq[c], false, bk1, (short)0, s1, false, false);
        }

        // --- tanh cap + masks + online softmax (per row, 16-lane reductions) ---
        const int kg0 = kbase + l16;
        const int kg1 = kbase + 16 + l16;
        const int am0 = amask[b * NL + kg0];
        const int am1 = amask[b * NL + kg1];

        #pragma unroll
        for (int r = 0; r < 8; ++r) {
            const int qg = qbase + r + half * 8;
            float v0 = TAU * tanhf(s0[r] * SCALE);
            float v1 = TAU * tanhf(s1[r] * SCALE);
            const bool ok0 = (kg0 <= qg) && (am0 != 0);
            const bool ok1 = (kg1 <= qg) && (am1 != 0);
            float c0 = ok0 ? v0 : -1e30f;
            float c1 = ok1 ? v1 : -1e30f;

            float t = fmaxf(c0, c1);
            #pragma unroll
            for (int off = 1; off < 16; off <<= 1)
                t = fmaxf(t, __shfl_xor(t, off, 32));
            const float m_new = fmaxf(m_run[r], t);
            const float alpha = __expf(m_run[r] - m_new);

            float p0 = ok0 ? __expf(v0 - m_new) : 0.0f;
            float p1 = ok1 ? __expf(v1 - m_new) : 0.0f;
            float rs = p0 + p1;
            #pragma unroll
            for (int off = 1; off < 16; off <<= 1)
                rs += __shfl_xor(rs, off, 32);

            l_run[r] = l_run[r] * alpha + rs;
            m_run[r] = m_new;
            #pragma unroll
            for (int c = 0; c < 8; ++c) o[c][r] *= alpha;

            // Scatter P into LDS row-major [16][32] (C-layout -> memory)
            const int row = r + half * 8;
            lds_p[wid][row * 32 + l16]      = (_Float16)p0;
            lds_p[wid][row * 32 + 16 + l16] = (_Float16)p1;
        }

        // --- gather P as A fragment (LDS in-order within wave) ---
        const _Float16* lp = &lds_p[wid][0];
        v8h alo = *(const v8h*)(lp + l16 * 32 + half * 16);
        v8h ahi = *(const v8h*)(lp + l16 * 32 + half * 16 + 8);
        v16h ap;
        #pragma unroll
        for (int i = 0; i < 8; ++i) { ap[i] = alo[i]; ap[8 + i] = ahi[i]; }

        // --- O += P @ V  (Vt is [B][DH][L]: 16 contiguous halves per lane) ---
        #pragma unroll
        for (int c = 0; c < 8; ++c) {
            const _Float16* vp =
                Vt + ((size_t)b * DH + c * 16 + l16) * NL + kbase + half * 16;
            v16h bv = *(const v16h*)(vp);
            o[c] = __builtin_amdgcn_wmma_f32_16x16x32_f16(
                false, ap, false, bv, (short)0, o[c], false, false);
        }
    }

    // --- normalize and write output (f32, [B][L][128]) ---
    float* op = out + ((size_t)b * NL + qbase) * DH;
    #pragma unroll
    for (int r = 0; r < 8; ++r) {
        const float inv = (l_run[r] > 0.0f) ? (1.0f / l_run[r]) : 0.0f;
        const int m = r + half * 8;
        #pragma unroll
        for (int c = 0; c < 8; ++c)
            op[(size_t)m * DH + c * 16 + l16] = o[c][r] * inv;
    }
}

// ---------------------------------------------------------------------------
extern "C" void kernel_launch(void* const* d_in, const int* in_sizes, int n_in,
                              void* d_out, int out_size, void* d_ws, size_t ws_size,
                              hipStream_t stream)
{
    (void)in_sizes; (void)n_in; (void)out_size; (void)ws_size;
    const float* x  = (const float*)d_in[0];
    const int*   am = (const int*)d_in[1];
    const float* wq = (const float*)d_in[2];
    const float* wk = (const float*)d_in[3];
    const float* wv = (const float*)d_in[4];
    float* out = (float*)d_out;

    char* ws = (char*)d_ws;
    const size_t qkv_bytes = (size_t)NB * NL * DH * sizeof(_Float16); // 4 MB
    _Float16* Qb = (_Float16*)(ws);
    _Float16* Kb = (_Float16*)(ws + qkv_bytes);
    _Float16* Vt = (_Float16*)(ws + 2 * qkv_bytes);

    // Projections: (B*L/16) * 3 blocks
    dim3 pgrid((NB * NL / 16) * 3);
    proj_kernel<<<pgrid, 256, 0, stream>>>(x, wq, wk, wv, Qb, Kb, Vt);

    // Attention: B*L/16 waves, 4 waves per block
    dim3 agrid((NB * NL / 16) / 4);
    attn_kernel<<<agrid, 128, 0, stream>>>(Qb, Kb, Vt, am, out);
}
